// Selector_7069516169879
// MI455X (gfx1250) — matrix operand order repaired and checked by reading, not compiled, not run on hardware
//
#include <hip/hip_runtime.h>
#include <hip/hip_bf16.h>
#include <stdint.h>

#define N_INST 262144
#define N_BAGS 16384
#define HIDDEN 768
#define REL_NUM 53
#define KK     (HIDDEN / 2)        /* 384 packed bf16-pairs per column */
#define NPAD   64                  /* 4 WMMA N-tiles of 16 */
#define WAVES  8                   /* waves per block */

typedef __bf16  v16bf __attribute__((ext_vector_type(16)));
typedef float   v8f   __attribute__((ext_vector_type(8)));
typedef float   f32x4 __attribute__((ext_vector_type(4)));
typedef uint32_t u32x4 __attribute__((ext_vector_type(4)));

#define NEG_INF (-__builtin_inff())

static __device__ inline uint32_t pk_bf16(float lo, float hi) {
    __bf16 l = (__bf16)lo, h = (__bf16)hi;
    unsigned short ls = __builtin_bit_cast(unsigned short, l);
    unsigned short hs = __builtin_bit_cast(unsigned short, h);
    return ((uint32_t)hs << 16) | (uint32_t)ls;
}

// Stage rel_mat [768 x 53] f32 -> LDS as packed bf16 pairs, layout Bpk[n][kk]
// (column n contiguous in kk so B fragments are two ds_load_b128).
static __device__ inline void stage_B(uint32_t* Bpk, const float* rel) {
    for (int i = threadIdx.x; i < NPAD * KK; i += blockDim.x) {
        int n = i / KK, kk = i - n * KK;
        float lo = 0.f, hi = 0.f;
        if (n < REL_NUM) {
            lo = rel[(2 * kk)     * REL_NUM + n];
            hi = rel[(2 * kk + 1) * REL_NUM + n];
        }
        Bpk[i] = pk_bf16(lo, hi);
    }
}

// A fragment, 16-bit 16x32 layout: lanes 0-15 K=base..base+7 & base+16..+23
// with base = 0 (lo half) / 8 (hi half); VGPR j holds K pair (2j, 2j+1).
// x is a single-use 805 MB stream -> non-temporal loads (keep L2 for B/out).
static __device__ inline v16bf load_a_frag(const float* xrow, int k0, int half) {
    const int kb = k0 + (half ? 8 : 0);
    const f32x4* p = reinterpret_cast<const f32x4*>(xrow + kb);
    const f32x4* q = reinterpret_cast<const f32x4*>(xrow + kb + 16);
    f32x4 f0 = __builtin_nontemporal_load(p);
    f32x4 f1 = __builtin_nontemporal_load(p + 1);
    f32x4 f2 = __builtin_nontemporal_load(q);
    f32x4 f3 = __builtin_nontemporal_load(q + 1);
    union { __bf16 e[16]; v16bf v; } a;
    a.e[0]  = (__bf16)f0.x; a.e[1]  = (__bf16)f0.y; a.e[2]  = (__bf16)f0.z; a.e[3]  = (__bf16)f0.w;
    a.e[4]  = (__bf16)f1.x; a.e[5]  = (__bf16)f1.y; a.e[6]  = (__bf16)f1.z; a.e[7]  = (__bf16)f1.w;
    a.e[8]  = (__bf16)f2.x; a.e[9]  = (__bf16)f2.y; a.e[10] = (__bf16)f2.z; a.e[11] = (__bf16)f2.w;
    a.e[12] = (__bf16)f3.x; a.e[13] = (__bf16)f3.y; a.e[14] = (__bf16)f3.z; a.e[15] = (__bf16)f3.w;
    return a.v;
}

// B fragment, 16-bit 32x16 layout: lanes 0-15 K=k0..k0+7 & k0+16..+23, lanes 16-31
// K=k0+8..k0+15 & k0+24..+31, VGPR j = K pair -> two 4-u32 runs.
static __device__ inline v16bf load_b_frag(const uint32_t* Bcol, int k0, int half) {
    const int kk = (k0 >> 1) + (half ? 4 : 0);
    const u32x4* p = reinterpret_cast<const u32x4*>(Bcol + kk);
    u32x4 a = p[0], b = p[2];
    union { uint32_t u[8]; v16bf v; } f;
    f.u[0] = a.x; f.u[1] = a.y; f.u[2] = a.z; f.u[3] = a.w;
    f.u[4] = b.x; f.u[5] = b.y; f.u[6] = b.z; f.u[7] = b.w;
    return f.v;
}

// One wave computes a 16x64 logit tile: acc[t][r] = logit(row rb+8*half+r, col t*16+n16)
static __device__ inline void gemm16x64(const float* xrow, const uint32_t* Bpk,
                                        int n16, int half, v8f acc[4]) {
    for (int k0 = 0; k0 < HIDDEN; k0 += 32) {
        v16bf a = load_a_frag(xrow, k0, half);
#pragma unroll
        for (int t = 0; t < 4; ++t) {
            v16bf b = load_b_frag(Bpk + (t * 16 + n16) * KK, k0, half);
            acc[t] = __builtin_amdgcn_wmma_f32_16x16x32_bf16(
                false, a, false, b, (short)0, acc[t], false, false);
        }
    }
}

// ---------------- K0: qrow[i] = query[bag(i)] --------------------------------
__global__ void qrow_kernel(const int* __restrict__ scope, const int* __restrict__ query,
                            int* __restrict__ qrow) {
    int b = blockIdx.x * blockDim.x + threadIdx.x;
    if (b >= N_BAGS) return;
    int s = scope[2 * b], e = scope[2 * b + 1], q = query[b];
    for (int i = s; i < e; ++i) qrow[i] = q;
}

// ---------------- K1: per-instance softmax score via WMMA --------------------
__global__ __launch_bounds__(256)
void score_kernel(const float* __restrict__ x, const float* __restrict__ rel,
                  const float* __restrict__ bias, const int* __restrict__ qrow,
                  float* __restrict__ score) {
    __shared__ uint32_t Bpk[NPAD * KK];
    stage_B(Bpk, rel);
    __syncthreads();

    const int l = threadIdx.x & 31, wave = threadIdx.x >> 5;
    const int n16 = l & 15, half = l >> 4;
    const int rowbase = (blockIdx.x * WAVES + wave) * 16;

    const float* xrow = x + (size_t)(rowbase + n16) * HIDDEN;  // A: M = lane%16
    v8f acc[4] = {};
    gemm16x64(xrow, Bpk, n16, half, acc);

    // add bias / mask padded columns
#pragma unroll
    for (int t = 0; t < 4; ++t) {
        int col = t * 16 + n16;
        if (col < REL_NUM) {
            float bv = bias[col];
#pragma unroll
            for (int r = 0; r < 8; ++r) acc[t][r] += bv;
        } else {
#pragma unroll
            for (int r = 0; r < 8; ++r) acc[t][r] = NEG_INF;
        }
    }

    // per-row softmax + gather at q, reduced across the 16-lane half
#pragma unroll
    for (int r = 0; r < 8; ++r) {
        const int row = rowbase + 8 * half + r;
        float m = fmaxf(fmaxf(acc[0][r], acc[1][r]), fmaxf(acc[2][r], acc[3][r]));
        for (int msk = 1; msk <= 8; msk <<= 1) m = fmaxf(m, __shfl_xor(m, msk, 32));

        float s = 0.f;
#pragma unroll
        for (int t = 0; t < 4; ++t)
            if (t * 16 + n16 < REL_NUM) s += __expf(acc[t][r] - m);
        for (int msk = 1; msk <= 8; msk <<= 1) s += __shfl_xor(s, msk, 32);

        const int q = qrow[row];
        float lq = NEG_INF;
#pragma unroll
        for (int t = 0; t < 4; ++t)
            if (t * 16 + n16 == q) lq = acc[t][r];
        for (int msk = 1; msk <= 8; msk <<= 1) lq = fmaxf(lq, __shfl_xor(lq, msk, 32));

        if (n16 == 0) score[row] = __expf(lq - m) / s;
    }
}

// ---------------- K2: per-bag argmax (last index among maxima) ---------------
__global__ void argmax_kernel(const int* __restrict__ scope, const float* __restrict__ score,
                              int* __restrict__ idx) {
    int b = blockIdx.x * blockDim.x + threadIdx.x;
    if (b >= N_BAGS) return;
    int s = scope[2 * b], e = scope[2 * b + 1];
    float best = NEG_INF;
    int bi = s;
    for (int i = s; i < e; ++i) {
        float v = score[i];
        if (v >= best) { best = v; bi = i; }   // '>=' -> keep the LAST max
    }
    idx[b] = bi;
}

// ---------------- K3: bag_logit = x[idx] @ rel + bias via WMMA ---------------
__global__ __launch_bounds__(256)
void baglogit_kernel(const float* __restrict__ x, const float* __restrict__ rel,
                     const float* __restrict__ bias, const int* __restrict__ idx,
                     float* __restrict__ out) {
    __shared__ uint32_t Bpk[NPAD * KK];
    stage_B(Bpk, rel);
    __syncthreads();

    const int l = threadIdx.x & 31, wave = threadIdx.x >> 5;
    const int n16 = l & 15, half = l >> 4;
    const int bagbase = (blockIdx.x * WAVES + wave) * 16;

    const int g = idx[bagbase + n16];                  // gathered instance row
    const float* xrow = x + (size_t)g * HIDDEN;
    v8f acc[4] = {};
    gemm16x64(xrow, Bpk, n16, half, acc);

#pragma unroll
    for (int t = 0; t < 4; ++t) {
        int col = t * 16 + n16;
        if (col < REL_NUM) {
            float bv = bias[col];
#pragma unroll
            for (int r = 0; r < 8; ++r) {
                int bag = bagbase + 8 * half + r;
                __builtin_nontemporal_store(acc[t][r] + bv,
                                            out + (size_t)bag * REL_NUM + col);
            }
        }
    }
}

extern "C" void kernel_launch(void* const* d_in, const int* in_sizes, int n_in,
                              void* d_out, int out_size, void* d_ws, size_t ws_size,
                              hipStream_t stream) {
    const float* x     = (const float*)d_in[0];
    const float* rel   = (const float*)d_in[1];
    const float* bias  = (const float*)d_in[2];
    const int*   scope = (const int*)d_in[3];
    const int*   query = (const int*)d_in[4];
    float*       out   = (float*)d_out;

    float* score = (float*)d_ws;                    // N_INST f32
    int*   qrow  = (int*)d_ws + N_INST;             // N_INST i32
    int*   idx   = (int*)d_ws + 2 * N_INST;         // N_BAGS i32

    qrow_kernel<<<(N_BAGS + 255) / 256, 256, 0, stream>>>(scope, query, qrow);
    score_kernel<<<N_INST / (16 * WAVES), 256, 0, stream>>>(x, rel, bias, qrow, score);
    argmax_kernel<<<(N_BAGS + 255) / 256, 256, 0, stream>>>(scope, score, idx);
    baglogit_kernel<<<N_BAGS / (16 * WAVES), 256, 0, stream>>>(x, rel, bias, idx, out);
}
